// AutoIntMLP_42202348651101
// MI455X (gfx1250) — matrix operand order repaired and checked by reading, not compiled, not run on hardware
//
#include <hip/hip_runtime.h>
#include <hip/hip_bf16.h>

#define BB 16384
#define FF 32
#define EE 64
#define AD 128
#define NH 2
#define DH 64

typedef __attribute__((ext_vector_type(16))) __bf16 v16bf;
typedef __attribute__((ext_vector_type(8)))  float  v8f;

struct alignas(16) Q16 { unsigned u[4]; };
struct alignas(16) BF16x16 { Q16 lo, hi; };

static __device__ inline unsigned short f2bf(float f) {
    unsigned u = __builtin_bit_cast(unsigned, f);
    u += 0x7FFFu + ((u >> 16) & 1u);
    return (unsigned short)(u >> 16);
}
static __device__ inline float bf2f(unsigned short h) {
    unsigned u = ((unsigned)h) << 16;
    return __builtin_bit_cast(float, u);
}

// A-operand (16x32 bf16): lane M=l%16; K(e) = e<8 ? 8*half+e : 16+8*half+(e-8)
static __device__ inline v16bf loadA(const unsigned short* rowp, int kbase, int half) {
    BF16x16 t;
    t.lo = *(const Q16*)(rowp + kbase + half * 8);
    t.hi = *(const Q16*)(rowp + kbase + 16 + half * 8);
    return __builtin_bit_cast(v16bf, t);
}
// B-operand (32x16 bf16): lane N=l%16; K(e) = e<8 ? 8*half+e : 16+8*half+(e-8)
static __device__ inline v16bf loadB(const unsigned short* rowp, int kbase, int half) {
    BF16x16 t;
    t.lo = *(const Q16*)(rowp + kbase + half * 8);
    t.hi = *(const Q16*)(rowp + kbase + 16 + half * 8);
    return __builtin_bit_cast(v16bf, t);
}

static __device__ inline v8f wmma_bf16(v16bf a, v16bf b, v8f c) {
    return __builtin_amdgcn_wmma_f32_16x16x32_bf16(false, a, false, b, (short)0, c,
                                                   false, false);
}

// ---------------- weight convert + transpose: W(K,N) f32 -> Wt(N,K) bf16 ---
__global__ void convert_transpose(const float* __restrict__ W,
                                  unsigned short* __restrict__ Wt,
                                  int Kd, int Nd) {
    int i = blockIdx.x * blockDim.x + threadIdx.x;
    if (i >= Kd * Nd) return;
    int n = i / Kd, k = i % Kd;
    Wt[i] = f2bf(W[(size_t)k * Nd + n]);
}

// ---------------- embedding gather: (B,F) idx -> (B,F,64) bf16 -------------
__global__ void embed_kernel(const int* __restrict__ x,
                             const float* __restrict__ emb,
                             unsigned short* __restrict__ out) {
    int i = blockIdx.x * blockDim.x + threadIdx.x;   // B*F*E = 33.5M < 2^31
    int e = i & (EE - 1);
    int bf = i >> 6;            // (b*F + f)
    int f = bf & (FF - 1);
    int row = x[bf] + f * 1000;
    out[i] = f2bf(emb[(size_t)row * EE + e]);
}

// ---------------- one MHSA layer: one sample per block (64 threads) --------
template <int DIN>
__global__ __launch_bounds__(64) void attn_layer(
    const unsigned short* __restrict__ tin,   // (B, 32, DIN) bf16
    const unsigned short* __restrict__ Wqt,   // (128, DIN) bf16 transposed
    const unsigned short* __restrict__ Wkt,
    const unsigned short* __restrict__ Wvt,
    const unsigned short* __restrict__ Wrt,
    unsigned short* __restrict__ tout)        // (B, 32, 128) bf16
{
    __shared__ __align__(16) unsigned short lds_t[FF][DIN];
    __shared__ __align__(16) unsigned short lds_q[FF][AD];
    __shared__ __align__(16) unsigned short lds_k[FF][AD];
    __shared__ __align__(16) unsigned short lds_vT[AD][FF];   // transposed v
    __shared__ __align__(16) float          lds_sc[NH][FF][FF];
    __shared__ __align__(16) unsigned short lds_att[NH][FF][FF];

    const int tid  = threadIdx.x;
    const int wave = tid >> 5;
    const int lane = tid & 31;
    const int half = lane >> 4;
    const int ln   = lane & 15;
    const int mbase = wave * 16;

    // 1) stage t tile into LDS
    {
        const Q16* src = (const Q16*)(tin + (size_t)blockIdx.x * FF * DIN);
        Q16* dst = (Q16*)(&lds_t[0][0]);
        const int nchunk = FF * DIN / 8;
        for (int i = tid; i < nchunk; i += 64) dst[i] = src[i];
    }
    __syncthreads();

    // 2) projections q,k,v,r = t @ W ; q/k row-major, v transposed, r in regs
    v8f racc[8];
    {
        const unsigned short* Wts[4] = { Wqt, Wkt, Wvt, Wrt };
        for (int p = 0; p < 4; ++p) {
            const unsigned short* Wt = Wts[p];
            v8f acc[8] = {};
            for (int kt = 0; kt < DIN / 32; ++kt) {
                v16bf a = loadA(&lds_t[mbase + ln][0], kt * 32, half);
#pragma unroll
                for (int nt = 0; nt < 8; ++nt) {
                    v16bf b = loadB(Wt + (size_t)(nt * 16 + ln) * DIN, kt * 32, half);
                    acc[nt] = wmma_bf16(a, b, acc[nt]);
                }
            }
#pragma unroll
            for (int nt = 0; nt < 8; ++nt) {
#pragma unroll
                for (int r0 = 0; r0 < 8; ++r0) {
                    int m = mbase + half * 8 + r0;
                    int n = nt * 16 + ln;
                    if (p == 3)      racc[nt][r0] = acc[nt][r0];
                    else if (p == 0) lds_q[m][n]  = f2bf(acc[nt][r0]);
                    else if (p == 1) lds_k[m][n]  = f2bf(acc[nt][r0]);
                    else             lds_vT[n][m] = f2bf(acc[nt][r0]);
                }
            }
        }
    }
    __syncthreads();

    // 3) scores = q . k^T per head (K = 64 dims)
    for (int hh = 0; hh < NH; ++hh) {
        v8f acc[2] = {};
        for (int kt = 0; kt < 2; ++kt) {
            v16bf a = loadA(&lds_q[mbase + ln][hh * DH], kt * 32, half);
#pragma unroll
            for (int nt = 0; nt < 2; ++nt) {
                v16bf b = loadB(&lds_k[nt * 16 + ln][hh * DH], kt * 32, half);
                acc[nt] = wmma_bf16(a, b, acc[nt]);
            }
        }
#pragma unroll
        for (int nt = 0; nt < 2; ++nt)
#pragma unroll
            for (int r0 = 0; r0 < 8; ++r0)
                lds_sc[hh][mbase + half * 8 + r0][nt * 16 + ln] = acc[nt][r0];
    }
    __syncthreads();

    // 4) softmax: 64 threads <-> 64 (head,query) rows
    {
        int hh = tid >> 5;
        int qr = tid & 31;
        const float* row = lds_sc[hh][qr];
        float mx = row[0];
#pragma unroll
        for (int k2 = 1; k2 < FF; ++k2) mx = fmaxf(mx, row[k2]);
        float ev[FF];
        float sum = 0.f;
#pragma unroll
        for (int k2 = 0; k2 < FF; ++k2) { ev[k2] = __expf(row[k2] - mx); sum += ev[k2]; }
        float inv = 1.0f / sum;
#pragma unroll
        for (int k2 = 0; k2 < FF; ++k2) lds_att[hh][qr][k2] = f2bf(ev[k2] * inv);
    }
    __syncthreads();

    // 5) out = relu(att @ v + r)
    {
        v8f acc[8] = {};
        for (int hh = 0; hh < NH; ++hh) {
            v16bf a = loadA(&lds_att[hh][mbase + ln][0], 0, half);  // K = 32 keys
#pragma unroll
            for (int ntl = 0; ntl < 4; ++ntl) {
                int nt = hh * 4 + ntl;
                v16bf b = loadB(&lds_vT[hh * DH + ntl * 16 + ln][0], 0, half);
                acc[nt] = wmma_bf16(a, b, acc[nt]);
            }
        }
        unsigned short* outp = tout + (size_t)blockIdx.x * FF * AD;
#pragma unroll
        for (int nt = 0; nt < 8; ++nt)
#pragma unroll
            for (int r0 = 0; r0 < 8; ++r0) {
                int m = mbase + half * 8 + r0;
                int n = nt * 16 + ln;
                float val = acc[nt][r0] + racc[nt][r0];
                outp[(size_t)m * AD + n] = f2bf(fmaxf(val, 0.0f));
            }
    }
}

// ---------------- C = relu(A @ Bt^T + bias), bf16 in/out --------------------
template <int K>
__global__ __launch_bounds__(128) void gemm_bias_relu(
    const unsigned short* __restrict__ A,   // (M, K) bf16
    const unsigned short* __restrict__ Bt,  // (N, K) bf16 (transposed weights)
    const float* __restrict__ bias,         // (N)
    unsigned short* __restrict__ C,         // (M, N) bf16
    int Ntotal)
{
    const int tid  = threadIdx.x;
    const int wave = tid >> 5;
    const int lane = tid & 31;
    const int half = lane >> 4;
    const int ln   = lane & 15;
    const int mbase = blockIdx.x * 16;
    const int nbase = blockIdx.y * 128 + wave * 32;

    const unsigned short* arow = A + (size_t)(mbase + ln) * K;
    v8f acc[2] = {};
    for (int kt = 0; kt < K / 32; ++kt) {
        v16bf a = loadA(arow, kt * 32, half);
#pragma unroll
        for (int nt = 0; nt < 2; ++nt) {
            v16bf b = loadB(Bt + (size_t)(nbase + nt * 16 + ln) * K, kt * 32, half);
            acc[nt] = wmma_bf16(a, b, acc[nt]);
        }
    }
#pragma unroll
    for (int nt = 0; nt < 2; ++nt) {
        float bv = bias[nbase + nt * 16 + ln];
#pragma unroll
        for (int r0 = 0; r0 < 8; ++r0) {
            int m = mbase + half * 8 + r0;
            int n = nbase + nt * 16 + ln;
            C[(size_t)m * Ntotal + n] = f2bf(fmaxf(acc[nt][r0] + bv, 0.0f));
        }
    }
}

// ---------- final head: sigmoid(att.wf + relu(h2.w3+b3)*sum(wf) + bf) ------
__global__ __launch_bounds__(256) void final_head(
    const unsigned short* __restrict__ att,  // (B, 4096) bf16
    const unsigned short* __restrict__ h2,   // (B, 256) bf16
    const float* __restrict__ w3, const float* __restrict__ b3,
    const float* __restrict__ wf, const float* __restrict__ bfs,
    float* __restrict__ out)
{
    const int wave = threadIdx.x >> 5;
    const int lane = threadIdx.x & 31;
    const size_t s = (size_t)blockIdx.x * 8 + wave;

    const unsigned short* ap = att + s * (FF * AD);
    float s1 = 0.f, swf = 0.f;
    for (int j = lane; j < FF * AD; j += 32) {
        float w = wf[j];
        s1  += bf2f(ap[j]) * w;
        swf += w;
    }
    const unsigned short* hp = h2 + s * 256;
    float s3 = 0.f;
    for (int j = lane; j < 256; j += 32) s3 += bf2f(hp[j]) * w3[j];

    for (int off = 16; off > 0; off >>= 1) {
        s1  += __shfl_xor(s1, off, 32);
        swf += __shfl_xor(swf, off, 32);
        s3  += __shfl_xor(s3, off, 32);
    }
    if (lane == 0) {
        float d3  = fmaxf(s3 + b3[0], 0.0f);
        float val = s1 + d3 * swf + bfs[0];
        out[s] = 1.0f / (1.0f + __expf(-val));
    }
}

extern "C" void kernel_launch(void* const* d_in, const int* in_sizes, int n_in,
                              void* d_out, int out_size, void* d_ws, size_t ws_size,
                              hipStream_t stream) {
    const int*   x   = (const int*)d_in[0];
    const float* emb = (const float*)d_in[1];
    const float* Wf32[12];
    for (int i = 0; i < 12; ++i) Wf32[i] = (const float*)d_in[2 + i];
    const float* w1 = (const float*)d_in[14];
    const float* b1 = (const float*)d_in[15];
    const float* w2 = (const float*)d_in[16];
    const float* b2 = (const float*)d_in[17];
    const float* w3 = (const float*)d_in[18];
    const float* b3 = (const float*)d_in[19];
    const float* wf = (const float*)d_in[20];
    const float* bf = (const float*)d_in[21];

    char* ws = (char*)d_ws;
    size_t off = 0;
    auto alloc = [&](size_t bytes) -> void* {
        void* p = ws + off;
        off += (bytes + 255) & ~(size_t)255;
        return p;
    };
    unsigned short* embed_bf = (unsigned short*)alloc((size_t)BB * FF * EE * 2);
    unsigned short* t1 = (unsigned short*)alloc((size_t)BB * FF * AD * 2);
    unsigned short* t2 = (unsigned short*)alloc((size_t)BB * FF * AD * 2);
    unsigned short* h1 = (unsigned short*)alloc((size_t)BB * 512 * 2);
    unsigned short* h2 = (unsigned short*)alloc((size_t)BB * 256 * 2);
    unsigned short* Wt[12];
    for (int l = 0; l < 3; ++l) {
        int din = (l == 0) ? EE : AD;
        for (int j = 0; j < 4; ++j)
            Wt[l * 4 + j] = (unsigned short*)alloc((size_t)din * AD * 2);
    }
    unsigned short* w1t = (unsigned short*)alloc((size_t)2048 * 512 * 2);
    unsigned short* w2t = (unsigned short*)alloc((size_t)512 * 256 * 2);

    // weight conversion (f32 -> transposed bf16)
    for (int l = 0; l < 3; ++l) {
        int din = (l == 0) ? EE : AD;
        for (int j = 0; j < 4; ++j) {
            int tot = din * AD;
            convert_transpose<<<(tot + 255) / 256, 256, 0, stream>>>(
                Wf32[l * 4 + j], Wt[l * 4 + j], din, AD);
        }
    }
    convert_transpose<<<(2048 * 512 + 255) / 256, 256, 0, stream>>>(w1, w1t, 2048, 512);
    convert_transpose<<<(512 * 256 + 255) / 256, 256, 0, stream>>>(w2, w2t, 512, 256);

    embed_kernel<<<(BB * FF * EE) / 256, 256, 0, stream>>>(x, emb, embed_bf);

    attn_layer<EE><<<BB, 64, 0, stream>>>(embed_bf, Wt[0], Wt[1], Wt[2], Wt[3], t1);
    attn_layer<AD><<<BB, 64, 0, stream>>>(t1, Wt[4], Wt[5], Wt[6], Wt[7], t2);
    attn_layer<AD><<<BB, 64, 0, stream>>>(t2, Wt[8], Wt[9], Wt[10], Wt[11], t1);

    gemm_bias_relu<2048><<<dim3(BB / 16, 4), 128, 0, stream>>>(embed_bf, w1t, b1, h1, 512);
    gemm_bias_relu<512><<<dim3(BB / 16, 2), 128, 0, stream>>>(h1, w2t, b2, h2, 256);

    final_head<<<BB / 8, 256, 0, stream>>>(t1, h2, w3, b3, wf, bf, (float*)d_out);
}
